// BioEncoder_27848567947265
// MI455X (gfx1250) — matrix operand order repaired and checked by reading, not compiled, not run on hardware
//
#include <hip/hip_runtime.h>
#include <hip/hip_bf16.h>
#include <math.h>

#define HIDF 128
#define EPSV 1e-5f

typedef __attribute__((ext_vector_type(16))) __bf16 v16bf;
typedef __attribute__((ext_vector_type(8)))  float  v8f;

union FragAB {
  v16bf v;
  uint4 q[2];
  unsigned short s[16];
};

__device__ __forceinline__ unsigned short f2bf(float f) {
  unsigned int u = __float_as_uint(f);
  u += 0x7FFFu + ((u >> 16) & 1u);   // round to nearest even
  return (unsigned short)(u >> 16);
}

// float atomic max via sign-aware integer atomics (init must be -inf bits)
__device__ __forceinline__ void atomicMaxF(float* addr, float val) {
  if (val >= 0.0f) atomicMax((int*)addr, __float_as_int(val));
  else             atomicMin((unsigned int*)addr, __float_as_uint(val));
}

// ---------------- generic fill ----------------
__global__ void k_fill_f32(float* __restrict__ p, float v, int n) {
  int i = blockIdx.x * blockDim.x + threadIdx.x;
  if (i < n) p[i] = v;
}

// ---------------- degree / norm ----------------
__global__ void k_deg_accum(const int* __restrict__ col, float* __restrict__ deg, int E) {
  int e = blockIdx.x * blockDim.x + threadIdx.x;
  if (e < E) atomicAdd(&deg[col[e]], 1.0f);
}

__global__ void k_deg_rsqrt(float* __restrict__ deg, int n) {
  int i = blockIdx.x * blockDim.x + threadIdx.x;
  if (i < n) {
    float d = deg[i];
    deg[i] = (d > 0.0f) ? rsqrtf(d) : 0.0f;
  }
}

// ---------------- f32 -> padded bf16 ----------------
__global__ void k_cvt_pad_bf16(const float* __restrict__ src, unsigned short* __restrict__ dst,
                               int M, int K, int Kp) {
  int idx = blockIdx.x * blockDim.x + threadIdx.x;
  if (idx >= M * Kp) return;
  int m = idx / Kp, k = idx - m * Kp;
  float v = (k < K) ? src[(size_t)m * K + k] : 0.0f;
  dst[idx] = f2bf(v);
}

// ---------------- pack W (K x 128, f32) into WMMA B-fragment layout ----------------
// Bp[((kt*8 + nt)*32 + lane)*16 + i] = B[kt*32 + (lane>>4)*16 + i, nt*16 + (lane&15)]
__global__ void k_pack_w(const float* __restrict__ W, unsigned short* __restrict__ Bp,
                         int K, int Kp) {
  int tid = blockIdx.x * blockDim.x + threadIdx.x;
  if (tid >= Kp * HIDF) return;
  int i    = tid & 15;
  int lane = (tid >> 4) & 31;
  int nt   = (tid >> 9) & 7;
  int kt   = tid >> 12;
  int kk = (kt << 5) + ((lane >> 4) << 4) + i;
  int n  = (nt << 4) + (lane & 15);
  float v = (kk < K) ? W[(size_t)kk * HIDF + n] : 0.0f;
  Bp[tid] = f2bf(v);
}

// ---------------- WMMA GEMM: C[M,128] = A[M,Kp](bf16) * Bpacked(bf16), opt bias/act ----
// block = 256 threads = 8 waves; wave w computes the 16x16 tile at columns [16w,16w+16)
// act: 0 none, 1 relu, 2 tanh
__global__ void k_gemm_wmma_bf16(const unsigned short* __restrict__ A,
                                 const unsigned short* __restrict__ Bp,
                                 const float* __restrict__ bias,
                                 float* __restrict__ C,
                                 int M, int Kp, int act) {
  const int wave = threadIdx.x >> 5;
  const int lane = threadIdx.x & 31;
  const int half = lane >> 4;
  const int lidx = lane & 15;
  const int m0 = blockIdx.x << 4;
  const int row = m0 + lidx;            // A row held by this lane
  v8f acc = {};
  const int KT = Kp >> 5;
  for (int kt = 0; kt < KT; ++kt) {
    FragAB a, b;
    // A 16x32 bf16 fragment: lanes 0-15 -> K 0..15 ; lanes 16-31 -> K 16..31
    const unsigned short* ap = A + (size_t)row * Kp + (kt << 5) + (half << 4);
    a.q[0] = *(const uint4*)(ap);
    a.q[1] = *(const uint4*)(ap + 8);
    // B fragment: pre-packed contiguous 32 bytes per lane
    const unsigned short* bp = Bp + (((size_t)(kt * 8 + wave) * 32 + lane) << 4);
    b.q[0] = *(const uint4*)(bp);
    b.q[1] = *(const uint4*)(bp + 8);
    acc = __builtin_amdgcn_wmma_f32_16x16x32_bf16(false, a.v, false, b.v,
                                                  (short)0, acc, false, false);
  }
  const int col = (wave << 4) + lidx;
  const float bv = bias ? bias[col] : 0.0f;
  #pragma unroll
  for (int j = 0; j < 8; ++j) {
    // D layout: lanes 0-15 VGPR j -> row m0+j ; lanes 16-31 -> row m0+8+j ; col = n0+lidx
    float v = acc[j] + bv;
    if (act == 1)      v = fmaxf(v, 0.0f);
    else if (act == 2) v = tanhf(v);
    C[(size_t)(m0 + (half << 3) + j) * HIDF + col] = v;
  }
}

// ---------------- GCN aggregation ----------------
// out[v,f] = bias[f] + xw[v,f]*dinv[v]^2   (self-loop term + bias init)
__global__ void k_init_agg(const float* __restrict__ xw, const float* __restrict__ dinv,
                           const float* __restrict__ bias, float* __restrict__ out, int N) {
  int idx = blockIdx.x * blockDim.x + threadIdx.x;
  if (idx >= N * HIDF) return;
  int v = idx >> 7, f = idx & 127;
  float d = dinv[v];
  out[idx] = bias[f] + xw[idx] * d * d;
}

// out[col,f] += xw[row,f] * dinv[row]*dinv[col]  for each edge
__global__ void k_scatter(const int* __restrict__ adj, const float* __restrict__ xw,
                          const float* __restrict__ dinv, float* __restrict__ out, int E) {
  int idx = blockIdx.x * blockDim.x + threadIdx.x;
  if (idx >= E * HIDF) return;
  int e = idx >> 7, f = idx & 127;
  int r = adj[e], c = adj[E + e];
  float w = dinv[r] * dinv[c];
  atomicAdd(&out[(size_t)c * HIDF + f], xw[(size_t)r * HIDF + f] * w);
}

// ---------------- batchnorm stats (biased, optional pre-relu) ----------------
__global__ void k_bn_stats(const float* __restrict__ x, int M, int doRelu,
                           float* __restrict__ s, float* __restrict__ sq) {
  int tid = blockIdx.x * blockDim.x + threadIdx.x;
  int col = tid & 127;
  int r0 = tid >> 7;
  int rs = (gridDim.x * blockDim.x) >> 7;
  float a = 0.0f, b = 0.0f;
  for (int r = r0; r < M; r += rs) {
    float v = x[(size_t)r * HIDF + col];
    if (doRelu) v = fmaxf(v, 0.0f);
    a += v; b += v * v;
  }
  atomicAdd(&s[col], a);
  atomicAdd(&sq[col], b);
}

// ---------------- batchnorm apply -> bf16 (feeds next WMMA GEMM) ----------------
__global__ void k_bn_apply_bf16(const float* __restrict__ x, int M,
                                const float* __restrict__ s, const float* __restrict__ sq,
                                const float* __restrict__ g, const float* __restrict__ beta,
                                int doRelu, unsigned short* __restrict__ out) {
  int idx = blockIdx.x * blockDim.x + threadIdx.x;
  if (idx >= M * HIDF) return;
  int col = idx & 127;
  float invM = 1.0f / (float)M;
  float m = s[col] * invM;
  float var = sq[col] * invM - m * m;
  float v = x[idx];
  if (doRelu) v = fmaxf(v, 0.0f);
  float y = (v - m) * rsqrtf(var + EPSV) * g[col] + beta[col];
  out[idx] = f2bf(y);
}

// ---------------- batchnorm apply + segment max pool ----------------
__global__ void k_bn_pool(const float* __restrict__ x, int M,
                          const float* __restrict__ s, const float* __restrict__ sq,
                          const float* __restrict__ g, const float* __restrict__ beta,
                          const int* __restrict__ ibatch, float* __restrict__ pool) {
  int idx = blockIdx.x * blockDim.x + threadIdx.x;
  if (idx >= M * HIDF) return;
  int r = idx >> 7, col = idx & 127;
  float invM = 1.0f / (float)M;
  float m = s[col] * invM;
  float var = sq[col] * invM - m * m;
  float v = fmaxf(x[idx], 0.0f);
  float y = (v - m) * rsqrtf(var + EPSV) * g[col] + beta[col];
  atomicMaxF(&pool[(size_t)ibatch[r] * HIDF + col], y);
}

static inline int cdiv(long long a, long long b) { return (int)((a + b - 1) / b); }

extern "C" void kernel_launch(void* const* d_in, const int* in_sizes, int n_in,
                              void* d_out, int out_size, void* d_ws, size_t ws_size,
                              hipStream_t stream) {
  const float* drug_feature = (const float*)d_in[0];
  const int*   drug_adj     = (const int*)d_in[1];
  const int*   ibatch       = (const int*)d_in[2];
  const float* gexpr        = (const float*)d_in[3];
  const float* W1  = (const float*)d_in[4];  const float* b1    = (const float*)d_in[5];
  const float* g1  = (const float*)d_in[6];  const float* beta1 = (const float*)d_in[7];
  const float* W2  = (const float*)d_in[8];  const float* b2    = (const float*)d_in[9];
  const float* g2  = (const float*)d_in[10]; const float* beta2 = (const float*)d_in[11];
  const float* Wc1 = (const float*)d_in[12]; const float* bc1   = (const float*)d_in[13];
  const float* gc1 = (const float*)d_in[14]; const float* betac1= (const float*)d_in[15];
  const float* Wc2 = (const float*)d_in[16]; const float* bc2   = (const float*)d_in[17];
  float* out = (float*)d_out;

  const int N = in_sizes[0] / 78;    // 200000
  const int E = in_sizes[1] / 2;     // 800000
  const int G = in_sizes[3] / 954;   // 8192
  const int K1p = 96, Kcp = 960;

  // workspace carve
  size_t off = 0;
  char* wsb = (char*)d_ws;
  auto carve = [&](size_t bytes) -> void* {
    void* p = wsb + off;
    off += (bytes + 255) & ~(size_t)255;
    return p;
  };
  float* deg            = (float*)carve((size_t)N * 4);            // becomes dinv
  float* xw             = (float*)carve((size_t)N * HIDF * 4);
  float* agg            = (float*)carve((size_t)N * HIDF * 4);
  unsigned short* a1bf  = (unsigned short*)carve((size_t)N * K1p * 2);
  unsigned short* x1bf  = (unsigned short*)carve((size_t)N * HIDF * 2);
  unsigned short* gexbf = (unsigned short*)carve((size_t)G * Kcp * 2);
  float* c1             = (float*)carve((size_t)G * HIDF * 4);
  unsigned short* c1nbf = (unsigned short*)carve((size_t)G * HIDF * 2);
  unsigned short* bp1   = (unsigned short*)carve((size_t)K1p * HIDF * 2);
  unsigned short* bp2   = (unsigned short*)carve((size_t)HIDF * HIDF * 2);
  unsigned short* bpc1  = (unsigned short*)carve((size_t)Kcp * HIDF * 2);
  unsigned short* bpc2  = (unsigned short*)carve((size_t)HIDF * HIDF * 2);
  float* stats          = (float*)carve(256 * 4);
  (void)ws_size; (void)n_in; (void)out_size;

  const int T = 256;

  // ---- degree / symmetric norm (self loop contributes 1 to every node) ----
  k_fill_f32<<<cdiv(N, T), T, 0, stream>>>(deg, 1.0f, N);
  k_deg_accum<<<cdiv(E, T), T, 0, stream>>>(drug_adj + E, deg, E);
  k_deg_rsqrt<<<cdiv(N, T), T, 0, stream>>>(deg, N);

  // ---- GCN layer 1: xw = drug_feature @ W1 (WMMA bf16) ----
  k_cvt_pad_bf16<<<cdiv((long long)N * K1p, T), T, 0, stream>>>(drug_feature, a1bf, N, 78, K1p);
  k_pack_w<<<cdiv(K1p * HIDF, T), T, 0, stream>>>(W1, bp1, 78, K1p);
  k_gemm_wmma_bf16<<<N / 16, T, 0, stream>>>(a1bf, bp1, nullptr, xw, N, K1p, 0);
  k_init_agg<<<cdiv((long long)N * HIDF, T), T, 0, stream>>>(xw, deg, b1, agg, N);
  k_scatter<<<cdiv((long long)E * HIDF, T), T, 0, stream>>>(drug_adj, xw, deg, agg, E);

  // ---- batchnorm(relu(x)) -> bf16 ----
  k_fill_f32<<<1, 256, 0, stream>>>(stats, 0.0f, 256);
  k_bn_stats<<<512, T, 0, stream>>>(agg, N, 1, stats, stats + 128);
  k_bn_apply_bf16<<<cdiv((long long)N * HIDF, T), T, 0, stream>>>(agg, N, stats, stats + 128,
                                                                  g1, beta1, 1, x1bf);

  // ---- GCN layer 2: xw = x1 @ W2 (WMMA bf16) ----
  k_pack_w<<<cdiv(HIDF * HIDF, T), T, 0, stream>>>(W2, bp2, HIDF, HIDF);
  k_gemm_wmma_bf16<<<N / 16, T, 0, stream>>>(x1bf, bp2, nullptr, xw, N, HIDF, 0);
  k_init_agg<<<cdiv((long long)N * HIDF, T), T, 0, stream>>>(xw, deg, b2, agg, N);
  k_scatter<<<cdiv((long long)E * HIDF, T), T, 0, stream>>>(drug_adj, xw, deg, agg, E);

  // ---- batchnorm(relu(x)) + global max pool -> out[0 : G*128) ----
  k_fill_f32<<<1, 256, 0, stream>>>(stats, 0.0f, 256);
  k_bn_stats<<<512, T, 0, stream>>>(agg, N, 1, stats, stats + 128);
  k_fill_f32<<<cdiv((long long)G * HIDF, T), T, 0, stream>>>(out, -__builtin_inff(), G * HIDF);
  k_bn_pool<<<cdiv((long long)N * HIDF, T), T, 0, stream>>>(agg, N, stats, stats + 128,
                                                            g2, beta2, ibatch, out);

  // ---- cell branch: c = tanh(gexpr @ Wc1 + bc1) (WMMA) ----
  k_cvt_pad_bf16<<<cdiv((long long)G * Kcp, T), T, 0, stream>>>(gexpr, gexbf, G, 954, Kcp);
  k_pack_w<<<cdiv(Kcp * HIDF, T), T, 0, stream>>>(Wc1, bpc1, 954, Kcp);
  k_gemm_wmma_bf16<<<G / 16, T, 0, stream>>>(gexbf, bpc1, bc1, c1, G, Kcp, 2);

  // ---- batchnorm(c) -> bf16 ----
  k_fill_f32<<<1, 256, 0, stream>>>(stats, 0.0f, 256);
  k_bn_stats<<<512, T, 0, stream>>>(c1, G, 0, stats, stats + 128);
  k_bn_apply_bf16<<<cdiv((long long)G * HIDF, T), T, 0, stream>>>(c1, G, stats, stats + 128,
                                                                  gc1, betac1, 0, c1nbf);

  // ---- relu(c @ Wc2 + bc2) -> out[G*128 : 2*G*128) (WMMA) ----
  k_pack_w<<<cdiv(HIDF * HIDF, T), T, 0, stream>>>(Wc2, bpc2, HIDF, HIDF);
  k_gemm_wmma_bf16<<<G / 16, T, 0, stream>>>(c1nbf, bpc2, bc2, out + (size_t)G * HIDF,
                                             G, HIDF, 1);
}
